// Attention_89799176225383
// MI455X (gfx1250) — matrix operand order repaired and checked by reading, not compiled, not run on hardware
//
#include <hip/hip_runtime.h>
#include <hip/hip_bf16.h>

typedef __bf16 bf16;
typedef __attribute__((ext_vector_type(16))) __bf16 v16bf;
typedef __attribute__((ext_vector_type(8)))  float  v8f;

#define B_  2
#define N_  2048
#define D_  768
#define H_  12
#define DH_ 64
#define IN_ 768
#define TRI 2304   // 3 * INNER

// ---------------- helpers ----------------

static __device__ __forceinline__ bf16 f2bf(float f) {
  union { float f; unsigned u; } c; c.f = f;
  unsigned u = c.u + 0x7FFFu + ((c.u >> 16) & 1u);   // round-to-nearest-even
  unsigned short h = (unsigned short)(u >> 16);
  bf16 r;
  __builtin_memcpy(&r, &h, 2);
  return r;
}

// Async 16-byte global -> LDS copy (per-lane addresses), tracked by ASYNCcnt.
// ISA 08_async_tensor: GV mode, LDS address taken from the VDST VGPR; the low 32 bits
// of a generic __shared__ pointer are exactly the wave-relative LDS offset.
static __device__ __forceinline__ void async_cp16(void* lds_dst, const void* gsrc) {
  unsigned lds_off = (unsigned)(size_t)lds_dst;
  asm volatile("global_load_async_to_lds_b128 %0, %1, off"
               :: "v"(lds_off), "v"(gsrc)
               : "memory");
}

// A-fragment (16x32 bf16). Lane L holds row L&15; lanes 0-15 K in
// 0..15, lanes 16-31 K in 16..31. Two 16B chunks per lane.
static __device__ __forceinline__ v16bf load_a_frag(const bf16* base, int ld) {
  const int lane = threadIdx.x & 31;
  const int row  = lane & 15;
  const int sel  = lane >> 4;
  const bf16* p0 = base + (size_t)row * ld + sel * 16;
  v16bf a;
  __builtin_memcpy(&a, p0, 16);
  __builtin_memcpy((char*)&a + 16, p0 + 8, 16);
  return a;
}

// B-fragment (32x16 bf16) from a K-transposed tile Bt[n][k] (k contiguous).
// Contiguous k order, as in load_a_frag: lanes 0-15 hold K=0..15 of column n=lane, lanes 16-31 K=16..31.
static __device__ __forceinline__ v16bf load_bt_frag(const bf16* base, int ld) {
  const int lane = threadIdx.x & 31;
  const int n    = lane & 15;
  const int kh   = lane >> 4;
  const bf16* p  = base + (size_t)n * ld + kh * 16;
  v16bf b;
  __builtin_memcpy(&b, p, 32);
  return b;
}

static __device__ __forceinline__ v8f wmma_bf16(v16bf a, v16bf b, v8f c) {
  return __builtin_amdgcn_wmma_f32_16x16x32_bf16(
      /*neg_a=*/false, a, /*neg_b=*/false, b,
      /*c_mod=*/(short)0, c, /*reuse_a=*/false, /*reuse_b=*/false);
}

// ---------------- kernel 1: QKV projection + layout scatter ----------------
// x[4096,768] @ w_qkv[768,2304] + b_qkv -> Q[b,h,n,d] (pre-scaled), K[b,h,n,d], Vt[b,h,d,n] (bf16)

__global__ __launch_bounds__(256)
void qkv_gemm_kernel(const float* __restrict__ x, const float* __restrict__ w,
                     const float* __restrict__ bias,
                     bf16* __restrict__ Q, bf16* __restrict__ K, bf16* __restrict__ Vt) {
  __shared__ bf16 As[128 * 40];   // [row][k], stride 40
  __shared__ bf16 Bs[128 * 40];   // [n][k] (transposed), stride 40

  const int tid   = threadIdx.x;
  const int mBase = blockIdx.y * 128;
  const int nBase = blockIdx.x * 128;
  const int wave  = tid >> 5;
  const int lane  = tid & 31;
  const int wm    = wave >> 1;   // 0..3 -> 32-row strip
  const int wn    = wave & 1;    // 0..1 -> 64-col strip

  v8f acc[2][4];
#pragma unroll
  for (int i = 0; i < 2; ++i)
#pragma unroll
    for (int j = 0; j < 4; ++j) acc[i][j] = (v8f){};

  for (int k0 = 0; k0 < D_; k0 += 32) {
    __syncthreads();
#pragma unroll
    for (int i = 0; i < 4; ++i) {
      int idx = tid + i * 256;
      int r = idx >> 3, c4 = idx & 7;
      const float4 v = *(const float4*)(x + (size_t)(mBase + r) * D_ + k0 + c4 * 4);
      bf16* dst = &As[r * 40 + c4 * 4];
      dst[0] = f2bf(v.x); dst[1] = f2bf(v.y); dst[2] = f2bf(v.z); dst[3] = f2bf(v.w);
    }
#pragma unroll
    for (int i = 0; i < 4; ++i) {
      int idx = tid + i * 256;
      int k = idx >> 5, n4 = idx & 31;
      const float4 v = *(const float4*)(w + (size_t)(k0 + k) * TRI + nBase + n4 * 4);
      Bs[(n4 * 4 + 0) * 40 + k] = f2bf(v.x);
      Bs[(n4 * 4 + 1) * 40 + k] = f2bf(v.y);
      Bs[(n4 * 4 + 2) * 40 + k] = f2bf(v.z);
      Bs[(n4 * 4 + 3) * 40 + k] = f2bf(v.w);
    }
    __syncthreads();

    v16bf a0 = load_a_frag(&As[(wm * 32 + 0) * 40], 40);
    v16bf a1 = load_a_frag(&As[(wm * 32 + 16) * 40], 40);
#pragma unroll
    for (int j = 0; j < 4; ++j) {
      v16bf bf = load_bt_frag(&Bs[(wn * 64 + j * 16) * 40], 40);
      acc[0][j] = wmma_bf16(a0, bf, acc[0][j]);
      acc[1][j] = wmma_bf16(a1, bf, acc[1][j]);
    }
  }

  const int half = lane >> 4;
  const int col0 = lane & 15;
#pragma unroll
  for (int i = 0; i < 2; ++i) {
#pragma unroll
    for (int j = 0; j < 4; ++j) {
#pragma unroll
      for (int s = 0; s < 8; ++s) {
        int m = mBase + wm * 32 + i * 16 + s + half * 8;
        int c = nBase + wn * 64 + j * 16 + col0;
        float v = acc[i][j][s] + bias[c];
        int mat = c / IN_;            // 0=q 1=k 2=v
        int ci  = c - mat * IN_;
        int h   = ci >> 6;
        int dh  = ci & 63;
        int b   = m >> 11;
        int n   = m & (N_ - 1);
        int bh  = b * H_ + h;
        if (mat == 0)      Q[((size_t)bh * N_ + n) * DH_ + dh] = f2bf(v * 0.125f); // fold 1/sqrt(64)
        else if (mat == 1) K[((size_t)bh * N_ + n) * DH_ + dh] = f2bf(v);
        else               Vt[((size_t)bh * DH_ + dh) * N_ + n] = f2bf(v);
      }
    }
  }
}

// ---------------- kernel 2: flash attention (async double-buffered K/V) ----------------

__global__ __launch_bounds__(256)
void attn_kernel(const bf16* __restrict__ Q, const bf16* __restrict__ K,
                 const bf16* __restrict__ Vt, bf16* __restrict__ AO) {
  __shared__ bf16 Ks[2][64 * 72];   // [key][d]   stride 72, ping-pong
  __shared__ bf16 Vs[2][64 * 72];   // [d][key]   stride 72, ping-pong
  __shared__ bf16 Ps[8][16 * 72];   // per-wave P staging, [qrow][key]

  const int tid  = threadIdx.x;
  const int wave = tid >> 5;
  const int lane = tid & 31;
  const int half = lane >> 4;
  const int col0 = lane & 15;

  const int bh    = blockIdx.y;
  const int qBase = blockIdx.x * 128 + wave * 16;

  const bf16* Qb = Q  + (size_t)bh * N_ * DH_;
  const bf16* Kb = K  + (size_t)bh * N_ * DH_;
  const bf16* Vb = Vt + (size_t)bh * DH_ * N_;

  // Q fragments held in registers for the whole kernel
  v16bf aq0 = load_a_frag(Qb + (size_t)qBase * DH_ + 0,  DH_);
  v16bf aq1 = load_a_frag(Qb + (size_t)qBase * DH_ + 32, DH_);

  v8f o[4];
#pragma unroll
  for (int t = 0; t < 4; ++t) o[t] = (v8f){};
  float m_run[8], l_run[8];
#pragma unroll
  for (int s = 0; s < 8; ++s) { m_run[s] = -3.0e38f; l_run[s] = 0.0f; }

  bf16* pw = &Ps[wave][0];

  // this thread's slice of a 64x64 tile: 32B of one row
  const int sr = tid >> 2;        // 0..63
  const int sc = tid & 3;         // 0..3 (16-element chunks)

  // prologue: prefetch key-block 0 into buffer 0 (4 async b128 per thread)
  {
    const bf16* gk = Kb + (size_t)sr * DH_ + sc * 16;
    const bf16* gv = Vb + (size_t)sr * N_ + sc * 16;
    async_cp16(&Ks[0][sr * 72 + sc * 16],     gk);
    async_cp16(&Ks[0][sr * 72 + sc * 16 + 8], gk + 8);
    async_cp16(&Vs[0][sr * 72 + sc * 16],     gv);
    async_cp16(&Vs[0][sr * 72 + sc * 16 + 8], gv + 8);
  }

  for (int kb = 0; kb < N_; kb += 64) {
    const int cur = (kb >> 6) & 1;
    const bool pre = (kb + 64) < N_;

    __syncthreads();   // everyone done reading the buffer we're about to refill
    if (pre) {
      const int nxt = cur ^ 1;
      const bf16* gk = Kb + (size_t)(kb + 64 + sr) * DH_ + sc * 16;
      const bf16* gv = Vb + (size_t)sr * N_ + (kb + 64) + sc * 16;
      async_cp16(&Ks[nxt][sr * 72 + sc * 16],     gk);
      async_cp16(&Ks[nxt][sr * 72 + sc * 16 + 8], gk + 8);
      async_cp16(&Vs[nxt][sr * 72 + sc * 16],     gv);
      async_cp16(&Vs[nxt][sr * 72 + sc * 16 + 8], gv + 8);
      // 4 newer ops may stay outstanding; in-order completion => current done
      asm volatile("s_wait_asynccnt 0x4" ::: "memory");
    } else {
      asm volatile("s_wait_asynccnt 0x0" ::: "memory");
    }
    __syncthreads();   // current buffer visible to the whole workgroup

    const bf16* ks = &Ks[cur][0];
    const bf16* vs = &Vs[cur][0];

    // S = (Q * scale) . K^T   (16 x 64 per wave)
    v8f s[4];
#pragma unroll
    for (int j = 0; j < 4; ++j) {
      v16bf bk0 = load_bt_frag(ks + (j * 16) * 72 + 0,  72);
      v16bf bk1 = load_bt_frag(ks + (j * 16) * 72 + 32, 72);
      v8f tmp = (v8f){};
      tmp  = wmma_bf16(aq0, bk0, tmp);
      s[j] = wmma_bf16(aq1, bk1, tmp);
    }

    // online softmax (row = slot + 8*half, cols across 16-lane group)
#pragma unroll
    for (int si = 0; si < 8; ++si) {
      float rm = fmaxf(fmaxf(s[0][si], s[1][si]), fmaxf(s[2][si], s[3][si]));
#pragma unroll
      for (int msk = 8; msk >= 1; msk >>= 1) rm = fmaxf(rm, __shfl_xor(rm, msk, 32));
      float mn = fmaxf(m_run[si], rm);
      float alpha = __expf(m_run[si] - mn);
      m_run[si] = mn;
      float rs = 0.0f;
#pragma unroll
      for (int j = 0; j < 4; ++j) {
        float p = __expf(s[j][si] - mn);
        s[j][si] = p;
        rs += p;
      }
#pragma unroll
      for (int msk = 8; msk >= 1; msk >>= 1) rs += __shfl_xor(rs, msk, 32);
      l_run[si] = l_run[si] * alpha + rs;
#pragma unroll
      for (int t = 0; t < 4; ++t) o[t][si] *= alpha;
    }

    // P: C-layout regs -> LDS (bf16), reload as A-fragments (wave-private)
#pragma unroll
    for (int j = 0; j < 4; ++j) {
#pragma unroll
      for (int si = 0; si < 8; ++si) {
        int row = si + half * 8;
        pw[row * 72 + j * 16 + col0] = f2bf(s[j][si]);
      }
    }
    v16bf ap0 = load_a_frag(pw + 0,  72);
    v16bf ap1 = load_a_frag(pw + 32, 72);

    // O += P . V   (V staged pre-transposed: Vs[d][key])
#pragma unroll
    for (int t = 0; t < 4; ++t) {
      v16bf bv0 = load_bt_frag(vs + (t * 16) * 72 + 0,  72);
      v16bf bv1 = load_bt_frag(vs + (t * 16) * 72 + 32, 72);
      o[t] = wmma_bf16(ap0, bv0, o[t]);
      o[t] = wmma_bf16(ap1, bv1, o[t]);
    }
  }

  // finalize: O /= l, write attn-out [b, n, h*64+d] bf16
  const int b = bh / H_;
  const int h = bh - b * H_;
#pragma unroll
  for (int si = 0; si < 8; ++si) {
    float inv = 1.0f / l_run[si];
    int n = qBase + si + 8 * half;
#pragma unroll
    for (int t = 0; t < 4; ++t) {
      float v = o[t][si] * inv;
      AO[((size_t)b * N_ + n) * IN_ + h * DH_ + t * 16 + col0] = f2bf(v);
    }
  }
}

// ---------------- kernel 3: output projection ----------------
// AO[4096,768] (bf16) @ w_out[768,768] + b_out -> out fp32

__global__ __launch_bounds__(256)
void out_gemm_kernel(const bf16* __restrict__ AO, const float* __restrict__ w,
                     const float* __restrict__ bias, float* __restrict__ out) {
  __shared__ bf16 As[128 * 40];
  __shared__ bf16 Bs[128 * 40];

  const int tid   = threadIdx.x;
  const int mBase = blockIdx.y * 128;
  const int nBase = blockIdx.x * 128;
  const int wave  = tid >> 5;
  const int lane  = tid & 31;
  const int wm    = wave >> 1;
  const int wn    = wave & 1;

  v8f acc[2][4];
#pragma unroll
  for (int i = 0; i < 2; ++i)
#pragma unroll
    for (int j = 0; j < 4; ++j) acc[i][j] = (v8f){};

  for (int k0 = 0; k0 < IN_; k0 += 32) {
    __syncthreads();
    {
      // A already bf16: async-copy while we convert B below
      int r = tid >> 1, ch = tid & 1;
      const bf16* g = AO + (size_t)(mBase + r) * IN_ + k0 + ch * 16;
      async_cp16(&As[r * 40 + ch * 16],     g);
      async_cp16(&As[r * 40 + ch * 16 + 8], g + 8);
    }
#pragma unroll
    for (int i = 0; i < 4; ++i) {
      int idx = tid + i * 256;
      int k = idx >> 5, n4 = idx & 31;
      const float4 v = *(const float4*)(w + (size_t)(k0 + k) * IN_ + nBase + n4 * 4);
      Bs[(n4 * 4 + 0) * 40 + k] = f2bf(v.x);
      Bs[(n4 * 4 + 1) * 40 + k] = f2bf(v.y);
      Bs[(n4 * 4 + 2) * 40 + k] = f2bf(v.z);
      Bs[(n4 * 4 + 3) * 40 + k] = f2bf(v.w);
    }
    asm volatile("s_wait_asynccnt 0x0" ::: "memory");
    __syncthreads();

    v16bf a0 = load_a_frag(&As[(wm * 32 + 0) * 40], 40);
    v16bf a1 = load_a_frag(&As[(wm * 32 + 16) * 40], 40);
#pragma unroll
    for (int j = 0; j < 4; ++j) {
      v16bf bf = load_bt_frag(&Bs[(wn * 64 + j * 16) * 40], 40);
      acc[0][j] = wmma_bf16(a0, bf, acc[0][j]);
      acc[1][j] = wmma_bf16(a1, bf, acc[1][j]);
    }
  }

  const int half = lane >> 4;
  const int col0 = lane & 15;
#pragma unroll
  for (int i = 0; i < 2; ++i) {
#pragma unroll
    for (int j = 0; j < 4; ++j) {
#pragma unroll
      for (int s = 0; s < 8; ++s) {
        int m = mBase + wm * 32 + i * 16 + s + half * 8;
        int c = nBase + wn * 64 + j * 16 + col0;
        out[(size_t)m * IN_ + c] = acc[i][j][s] + bias[c];
      }
    }
  }
}

// ---------------- launcher ----------------

extern "C" void kernel_launch(void* const* d_in, const int* in_sizes, int n_in,
                              void* d_out, int out_size, void* d_ws, size_t ws_size,
                              hipStream_t stream) {
  (void)in_sizes; (void)n_in; (void)out_size; (void)ws_size;
  const float* x     = (const float*)d_in[0];
  const float* w_qkv = (const float*)d_in[1];
  const float* b_qkv = (const float*)d_in[2];
  const float* w_out = (const float*)d_in[3];
  const float* b_out = (const float*)d_in[4];
  float* out = (float*)d_out;

  const size_t qkvElems = (size_t)B_ * H_ * N_ * DH_;   // 3,145,728
  char* ws = (char*)d_ws;
  bf16* Q  = (bf16*)ws;
  bf16* K  = Q  + qkvElems;
  bf16* Vt = K  + qkvElems;
  bf16* AO = Vt + qkvElems;                             // [B,N,INNER] bf16

  qkv_gemm_kernel<<<dim3(TRI / 128, (B_ * N_) / 128), 256, 0, stream>>>(
      x, w_qkv, b_qkv, Q, K, Vt);
  attn_kernel<<<dim3(N_ / 128, B_ * H_), 256, 0, stream>>>(Q, K, Vt, AO);
  out_gemm_kernel<<<dim3(IN_ / 128, (B_ * N_) / 128), 256, 0, stream>>>(
      AO, w_out, b_out, out);
}
